// SelfAttention_53377853554817
// MI455X (gfx1250) — matrix operand order repaired and checked by reading, not compiled, not run on hardware
//
#include <hip/hip_runtime.h>
#include <hip/hip_bf16.h>
#include <stdint.h>

typedef __bf16 bf16;
typedef __attribute__((ext_vector_type(16))) __bf16 v16bf;
typedef __attribute__((ext_vector_type(8)))  float  v8f;

#define DMODEL 4096
#define NHEADS 32
#define DHEAD  128
#define BATCH  2
#define SEQT   2048
#define LOG2E  1.4426950408889634f

union Frag { v16bf v; uint4 u[2]; };

static __device__ inline v8f zero8() {
  v8f z = {0.f,0.f,0.f,0.f,0.f,0.f,0.f,0.f};
  return z;
}

// Async global -> LDS copy of 16B per lane (CDNA5 GLOBAL_LOAD_ASYNC_TO_LDS_B128,
// tracked by ASYNCcnt). ldsAddr is the wave-relative LDS byte address = low 32
// bits of the flat address of a __shared__ object (ISA 10.2 aperture mapping).
static __device__ inline void asyncCopy16(uint32_t ldsAddr, const void* gaddr) {
  asm volatile("global_load_async_to_lds_b128 %0, %1, off"
               :: "v"(ldsAddr), "v"(gaddr) : "memory");
}
static __device__ inline void waitAsync0() {
  asm volatile("s_wait_asynccnt 0x0" ::: "memory");
}
static __device__ inline uint32_t ldsAddrOf(const void* p) {
  return (uint32_t)(uintptr_t)p;
}

// ---------------------------------------------------------------------------
// fp32 -> bf16 conversion, 8 elements / thread (b128 loads + b128 store)
// ---------------------------------------------------------------------------
__global__ __launch_bounds__(256) void cvt_f32_bf16(const float* __restrict__ in,
                                                    bf16* __restrict__ out, int n8) {
  int i = blockIdx.x * blockDim.x + threadIdx.x;
  if (i >= n8) return;
  const float4* p = (const float4*)in + (size_t)i * 2;
  float4 a = p[0], b = p[1];
  union { bf16 h[8]; uint4 u; } pk;
  pk.h[0] = (bf16)a.x; pk.h[1] = (bf16)a.y; pk.h[2] = (bf16)a.z; pk.h[3] = (bf16)a.w;
  pk.h[4] = (bf16)b.x; pk.h[5] = (bf16)b.y; pk.h[6] = (bf16)b.z; pk.h[7] = (bf16)b.w;
  *(uint4*)(out + (size_t)i * 8) = pk.u;
}

// ---------------------------------------------------------------------------
// bf16 WMMA GEMM:  C[M,N] = A[M,K] * B[N,K]^T   (einsum 'mk,nk->mn')
// block = 256 threads = 8 waves, block tile 128x256, K-step 32.
// Double-buffered LDS filled by async global->LDS b128 (no VGPR transit);
// one s_wait_asynccnt + one barrier per K-step. Wave tile 64x64 = 4x4 WMMAs.
// ---------------------------------------------------------------------------
#define BM 128
#define BN 256
#define BK 32
#define BKP 40   // +8 halves pad: LDS row stride 80B kills b128 bank conflicts

__device__ inline void storeOut(float* p, float v) { *p = v; }
__device__ inline void storeOut(bf16*  p, float v) { *p = (bf16)v; }

template <typename OutT>
__global__ __launch_bounds__(256) void gemm_bf16_nt(const bf16* __restrict__ A,
                                                    const bf16* __restrict__ Bw,
                                                    OutT* __restrict__ C,
                                                    int M, int N, int K) {
  __shared__ __align__(16) bf16 As[2][BM][BKP];
  __shared__ __align__(16) bf16 Bs[2][BN][BKP];

  const int tid  = threadIdx.x;
  const int lane = tid & 31;
  const int wave = tid >> 5;
  const int lr = lane & 15, hi = lane >> 4;
  const int wr = (wave & 1) * 64;   // wave row offset in tile
  const int wc = (wave >> 1) * 64;  // wave col offset in tile
  const int m0 = blockIdx.y * BM;
  const int n0 = blockIdx.x * BN;

  // Async stage one K-step: A tile = 512 16B chunks, B tile = 1024 16B chunks.
  auto stage = [&](int p, int k0) {
#pragma unroll
    for (int i = 0; i < 2; ++i) {               // A: 2 chunks / thread
      int c = tid + i * 256;
      int row = c >> 2, off = (c & 3) * 16;
      const char* g = (const char*)(A + (size_t)(m0 + row) * K + k0) + off;
      asyncCopy16(ldsAddrOf((const char*)&As[p][row][0] + off), g);
    }
#pragma unroll
    for (int i = 0; i < 4; ++i) {               // B: 4 chunks / thread
      int c = tid + i * 256;
      int row = c >> 2, off = (c & 3) * 16;
      const char* g = (const char*)(Bw + (size_t)(n0 + row) * K + k0) + off;
      asyncCopy16(ldsAddrOf((const char*)&Bs[p][row][0] + off), g);
    }
  };

  v8f acc[4][4];
#pragma unroll
  for (int i = 0; i < 4; ++i)
#pragma unroll
    for (int j = 0; j < 4; ++j) acc[i][j] = zero8();

  stage(0, 0);
  int p = 0;
#pragma unroll 1
  for (int k0 = 0; k0 < K; k0 += BK) {
    waitAsync0();          // buffer p fully written (own ops; barrier covers others)
    __syncthreads();
    if (k0 + BK < K) stage(p ^ 1, k0 + BK);     // overlap next transfer w/ compute

    Frag af[4], bfr[4];
#pragma unroll
    for (int i = 0; i < 4; ++i) {            // A 16x32: lane=M, K chunks 8*hi / 16+8*hi
      int row = wr + i * 16 + lr;
      af[i].u[0] = *(const uint4*)&As[p][row][8 * hi];
      af[i].u[1] = *(const uint4*)&As[p][row][16 + 8 * hi];
    }
#pragma unroll
    for (int j = 0; j < 4; ++j) {            // B 32x16: lane=N, K chunks 8*hi / 16+8*hi
      int row = wc + j * 16 + lr;
      bfr[j].u[0] = *(const uint4*)&Bs[p][row][8 * hi];
      bfr[j].u[1] = *(const uint4*)&Bs[p][row][16 + 8 * hi];
    }
#pragma unroll
    for (int i = 0; i < 4; ++i)
#pragma unroll
      for (int j = 0; j < 4; ++j)
        acc[i][j] = __builtin_amdgcn_wmma_f32_16x16x32_bf16(
            false, af[i].v, false, bfr[j].v, (short)0, acc[i][j], false, false);
    p ^= 1;
  }

  // C/D layout: VGPR r -> row = r + 8*hi, col = lane&15
#pragma unroll
  for (int i = 0; i < 4; ++i)
#pragma unroll
    for (int j = 0; j < 4; ++j)
#pragma unroll
      for (int r = 0; r < 8; ++r) {
        int row = m0 + wr + i * 16 + r + 8 * hi;
        int col = n0 + wc + j * 16 + lr;
        storeOut(C + (size_t)row * N + col, acc[i][j][r]);
      }
}

// ---------------------------------------------------------------------------
// RoPE on Q,K + relayout to [B,H,T,Dh]; V transposed to [B,H,Dh,T]
// ---------------------------------------------------------------------------
__global__ __launch_bounds__(256) void rope_reshape(const bf16* __restrict__ qp,
                                                    const bf16* __restrict__ kp,
                                                    const bf16* __restrict__ vp,
                                                    bf16* __restrict__ q,
                                                    bf16* __restrict__ k,
                                                    bf16* __restrict__ vt) {
  long p = (long)blockIdx.x * blockDim.x + threadIdx.x;
  int i = (int)(p & 63);        // rotary pair index 0..63
  long rest = p >> 6;
  int t = (int)(rest & (SEQT - 1));
  rest >>= 11;
  int h = (int)(rest & (NHEADS - 1));
  int b = (int)(rest >> 5);

  size_t src = ((size_t)b * SEQT + t) * DMODEL + (size_t)h * DHEAD + 2 * i;
  float x1q = (float)qp[src], x2q = (float)qp[src + 1];
  float x1k = (float)kp[src], x2k = (float)kp[src + 1];

  // inv_freq = theta^(-(2i)/128) = exp(-(2i/128)*ln(10000))
  float inv = __expf(-((float)(2 * i) * (1.0f / 128.0f)) * 9.210340371976184f);
  float ang = (float)t * inv;
  float s, c;
  __sincosf(ang, &s, &c);

  size_t dst = (((size_t)b * NHEADS + h) * SEQT + t) * DHEAD + 2 * i;
  q[dst]     = (bf16)(x1q * c - x2q * s);
  q[dst + 1] = (bf16)(x1q * s + x2q * c);
  k[dst]     = (bf16)(x1k * c - x2k * s);
  k[dst + 1] = (bf16)(x1k * s + x2k * c);

  size_t vb = ((size_t)b * NHEADS + h) * (size_t)DHEAD * SEQT;
  vt[vb + (size_t)(2 * i) * SEQT + t]     = vp[src];
  vt[vb + (size_t)(2 * i + 1) * SEQT + t] = vp[src + 1];
}

// ---------------------------------------------------------------------------
// Flash attention: block = 4 waves, each wave owns a 16-row Q tile.
// K (64x128) and V^T (128x64) tiles double-buffered in LDS, filled by async
// global->LDS b128 (shared by the 4 waves -> 4x less VMEM). Per 64-key step:
// 16 WMMA scores + one online-softmax pass + LDS P transpose + 16 WMMA P*V.
// ---------------------------------------------------------------------------
__global__ __launch_bounds__(128) void flash_attn(const bf16* __restrict__ Q,
                                                  const bf16* __restrict__ K,
                                                  const bf16* __restrict__ Vt,
                                                  bf16* __restrict__ O) {
  __shared__ __align__(16) bf16 Klds[2][64][136];  // 272B row stride (pad 8)
  __shared__ __align__(16) bf16 Vlds[2][128][72];  // 144B row stride (pad 8)
  __shared__ __align__(16) bf16 Plds[4][16][72];   // per-wave P transpose buffer

  const int tid  = threadIdx.x;
  const int lane = tid & 31;
  const int wave = tid >> 5;
  const int lr = lane & 15, hi = lane >> 4;
  const int b = blockIdx.z, h = blockIdx.y;
  const int q0 = blockIdx.x * 64 + wave * 16;

  const size_t bh = (size_t)b * NHEADS + h;
  const bf16* Qb = Q  + (bh * SEQT + q0) * DHEAD;
  const bf16* Kb = K  + bh * SEQT * DHEAD;
  const bf16* Vb = Vt + bh * (size_t)DHEAD * SEQT;

  // stage K tile: 64 rows x 256B = 1024 chunks; V tile: 128 rows x 128B = 1024.
  auto stageKV = [&](int p, int kt) {
#pragma unroll
    for (int i = 0; i < 8; ++i) {
      int c = tid + i * 128;
      int row = c >> 4, off = (c & 15) * 16;
      const char* g = (const char*)(Kb + (size_t)(kt * 64 + row) * DHEAD) + off;
      asyncCopy16(ldsAddrOf((const char*)&Klds[p][row][0] + off), g);
    }
#pragma unroll
    for (int i = 0; i < 8; ++i) {
      int c = tid + i * 128;
      int row = c >> 3, off = (c & 7) * 16;
      const char* g = (const char*)(Vb + (size_t)row * SEQT + kt * 64) + off;
      asyncCopy16(ldsAddrOf((const char*)&Vlds[p][row][0] + off), g);
    }
  };

  // Q A-fragments: 16x128 split into 4 chunks of K=32, held for the whole loop
  Frag qa[4];
#pragma unroll
  for (int c = 0; c < 4; ++c) {
    const bf16* base = Qb + (size_t)lr * DHEAD + 32 * c + 8 * hi;
    qa[c].u[0] = *(const uint4*)base;
    qa[c].u[1] = *(const uint4*)(base + 16);
  }

  v8f o[8];
#pragma unroll
  for (int t = 0; t < 8; ++t) o[t] = zero8();
  float mrow[8], lrow[8];
#pragma unroll
  for (int r = 0; r < 8; ++r) { mrow[r] = -1e30f; lrow[r] = 0.f; }

  const float scale = 0.08838834764831845f;   // 1/sqrt(128)
  const int NT = SEQT / 64;

  stageKV(0, 0);
  int p = 0;
#pragma unroll 1
  for (int kt = 0; kt < NT; ++kt) {
    waitAsync0();          // own async ops for buffer p complete
    __syncthreads();       // everyone's writes visible; prev reads of p^1 drained
    if (kt + 1 < NT) stageKV(p ^ 1, kt + 1);   // overlap next tile transfer

    v8f s[4];
#pragma unroll
    for (int g = 0; g < 4; ++g) s[g] = zero8();
#pragma unroll
    for (int c = 0; c < 4; ++c) {
#pragma unroll
      for (int g = 0; g < 4; ++g) {
        // B frag for S = Q*K^T: element (kdim, n) = K[key n][dh kdim]
        Frag bk;
        bk.u[0] = *(const uint4*)&Klds[p][g * 16 + lr][32 * c + 8 * hi];
        bk.u[1] = *(const uint4*)&Klds[p][g * 16 + lr][32 * c + 16 + 8 * hi];
        s[g] = __builtin_amdgcn_wmma_f32_16x16x32_bf16(false, qa[c].v, false, bk.v,
                                                       (short)0, s[g], false, false);
      }
    }

    float alpha[8];
#pragma unroll
    for (int r = 0; r < 8; ++r) {
      float v0 = s[0][r] * scale, v1 = s[1][r] * scale;
      float v2 = s[2][r] * scale, v3 = s[3][r] * scale;
      float mx = fmaxf(fmaxf(v0, v1), fmaxf(v2, v3));
#pragma unroll
      for (int d = 1; d < 16; d <<= 1) mx = fmaxf(mx, __shfl_xor(mx, d, 32));
      float mn = fmaxf(mrow[r], mx);
      float a  = exp2f((mrow[r] - mn) * LOG2E);
      float p0 = exp2f((v0 - mn) * LOG2E);
      float p1 = exp2f((v1 - mn) * LOG2E);
      float p2 = exp2f((v2 - mn) * LOG2E);
      float p3 = exp2f((v3 - mn) * LOG2E);
      float rs = (p0 + p1) + (p2 + p3);
#pragma unroll
      for (int d = 1; d < 16; d <<= 1) rs += __shfl_xor(rs, d, 32);
      lrow[r]  = lrow[r] * a + rs;
      mrow[r]  = mn;
      alpha[r] = a;
      int m = r + 8 * hi;                    // C/D-layout row
      Plds[wave][m][lr]      = (bf16)p0;     // keys  0..15
      Plds[wave][m][16 + lr] = (bf16)p1;     // keys 16..31
      Plds[wave][m][32 + lr] = (bf16)p2;     // keys 32..47
      Plds[wave][m][48 + lr] = (bf16)p3;     // keys 48..63
    }
#pragma unroll
    for (int t = 0; t < 8; ++t)
#pragma unroll
      for (int r = 0; r < 8; ++r) o[t][r] *= alpha[r];

    __syncthreads();       // P visible; next P store is after next top barrier
    Frag pa[2];            // re-read P as two A 16x32 fragments
#pragma unroll
    for (int kc = 0; kc < 2; ++kc) {
      const bf16* pr = &Plds[wave][lr][kc * 32 + 8 * hi];
      pa[kc].u[0] = *(const uint4*)pr;
      pa[kc].u[1] = *(const uint4*)(pr + 16);
    }

#pragma unroll
    for (int t = 0; t < 8; ++t) {
#pragma unroll
      for (int kc = 0; kc < 2; ++kc) {
        // B frag for O += P*V: element (k, n) = V[key k][dh n] = Vt[dh n][key k]
        Frag bv;
        bv.u[0] = *(const uint4*)&Vlds[p][t * 16 + lr][kc * 32 + 8 * hi];
        bv.u[1] = *(const uint4*)&Vlds[p][t * 16 + lr][kc * 32 + 16 + 8 * hi];
        o[t] = __builtin_amdgcn_wmma_f32_16x16x32_bf16(false, pa[kc].v, false, bv.v,
                                                       (short)0, o[t], false, false);
      }
    }
    p ^= 1;
  }

  float inv[8];
#pragma unroll
  for (int r = 0; r < 8; ++r) inv[r] = 1.0f / lrow[r];
#pragma unroll
  for (int t = 0; t < 8; ++t)
#pragma unroll
    for (int r = 0; r < 8; ++r) {
      int row = q0 + r + 8 * hi;
      O[((size_t)b * SEQT + row) * DMODEL + (size_t)h * DHEAD + t * 16 + lr] =
          (bf16)(o[t][r] * inv[r]);
    }
}

// ---------------------------------------------------------------------------
// Host-side launch
// ---------------------------------------------------------------------------
extern "C" void kernel_launch(void* const* d_in, const int* in_sizes, int n_in,
                              void* d_out, int out_size, void* d_ws, size_t ws_size,
                              hipStream_t stream) {
  (void)in_sizes; (void)n_in; (void)out_size; (void)ws_size;
  const float* hs = (const float*)d_in[0];
  const float* Wq = (const float*)d_in[1];
  const float* Wk = (const float*)d_in[2];
  const float* Wv = (const float*)d_in[3];
  const float* Wo = (const float*)d_in[4];
  float* out = (float*)d_out;

  const size_t BT = (size_t)BATCH * SEQT;     // 4096 rows
  const size_t CC = (size_t)DMODEL * DMODEL;  // weight elements
  const size_t EL = BT * DMODEL;              // activation elements

  bf16* p   = (bf16*)d_ws;
  bf16* hsb = p; p += EL;
  bf16* wqb = p; p += CC;
  bf16* wkb = p; p += CC;
  bf16* wvb = p; p += CC;
  bf16* wob = p; p += CC;
  bf16* qp  = p; p += EL;
  bf16* kp  = p; p += EL;
  bf16* vp  = p; p += EL;
  bf16* qr  = p; p += EL;
  bf16* kr  = p; p += EL;
  bf16* vt  = p; p += EL;
  bf16* at  = p; p += EL;

  int a8 = (int)(EL / 8), w8 = (int)(CC / 8);
  cvt_f32_bf16<<<(a8 + 255) / 256, 256, 0, stream>>>(hs, hsb, a8);
  cvt_f32_bf16<<<(w8 + 255) / 256, 256, 0, stream>>>(Wq, wqb, w8);
  cvt_f32_bf16<<<(w8 + 255) / 256, 256, 0, stream>>>(Wk, wkb, w8);
  cvt_f32_bf16<<<(w8 + 255) / 256, 256, 0, stream>>>(Wv, wvb, w8);
  cvt_f32_bf16<<<(w8 + 255) / 256, 256, 0, stream>>>(Wo, wob, w8);

  dim3 gg(DMODEL / BN, (unsigned)(BT / BM));
  gemm_bf16_nt<bf16><<<gg, 256, 0, stream>>>(hsb, wqb, qp, (int)BT, DMODEL, DMODEL);
  gemm_bf16_nt<bf16><<<gg, 256, 0, stream>>>(hsb, wkb, kp, (int)BT, DMODEL, DMODEL);
  gemm_bf16_nt<bf16><<<gg, 256, 0, stream>>>(hsb, wvb, vp, (int)BT, DMODEL, DMODEL);

  long pairs = (long)BATCH * NHEADS * SEQT * 64;
  rope_reshape<<<(unsigned)(pairs / 256), 256, 0, stream>>>(qp, kp, vp, qr, kr, vt);

  dim3 ga(SEQT / 64, NHEADS, BATCH);
  flash_attn<<<ga, 128, 0, stream>>>(qr, kr, vt, at);

  gemm_bf16_nt<float><<<gg, 256, 0, stream>>>(at, wob, out, (int)BT, DMODEL, DMODEL);
}
